// VTNLongformerModel_79190607003670
// MI455X (gfx1250) — matrix operand order repaired and checked by reading, not compiled, not run on hardware
//
#include <hip/hip_runtime.h>
#include <hip/hip_bf16.h>

// Longformer dims
#define BB  2
#define SS  4096
#define DD  768
#define HH  12
#define LL  3
#define FF  3072
#define WW  128
#define DHD 64
#define NCH 32
#define TT  (BB*SS)

typedef __attribute__((ext_vector_type(16))) __bf16 v16bf;
typedef __attribute__((ext_vector_type(8)))  float  v8f;

__device__ __forceinline__ int imin(int a, int b) { return a < b ? a : b; }
__device__ __forceinline__ int imax(int a, int b) { return a > b ? a : b; }

// Async global->LDS DMA (VGLOBAL GLOBAL_LOAD_ASYNC_TO_LDS_B128, ASYNCcnt).
// ldsoff = LDS byte offset (low 32 bits of flat shared address), ga = global addr.
__device__ __forceinline__ void async_ld_lds_b128(unsigned ldsoff,
                                                  unsigned long long ga) {
  asm volatile("global_load_async_to_lds_b128 %0, %1, off"
               :: "v"(ldsoff), "v"(ga)
               : "memory");
}
__device__ __forceinline__ void wait_async0() {
  asm volatile("s_wait_asynccnt 0x0" ::: "memory");
}
__device__ __forceinline__ unsigned lds_off(const void* p) {
  return (unsigned)(uintptr_t)p;  // LDS aperture: low 32 bits carry the offset
}

// --- CDNA5 WMMA fragment loaders (contiguous K = kh*16+i order, same for A and B, wave32) ----------------
// A (16x32 bf16): lane m=lane&15 is row; elems 0..15 = K kh*16..kh*16+15
//                 (kh = lane>>4)
__device__ __forceinline__ v16bf load_a_frag(const __bf16* p, int lda, int lane) {
  int m = lane & 15, kh = lane >> 4;
  const __bf16* r = p + (size_t)m * lda;
  int k0 = kh * 16;
  v16bf a;
#pragma unroll
  for (int i = 0; i < 16; ++i) a[i] = r[k0 + i];
  return a;
}
// B (32x16 bf16): lane n=lane&15 is column; elems 0..15 = K kh*16..kh*16+15.
// p points to column-major storage: p[n*ldb + k]
__device__ __forceinline__ v16bf load_b_frag(const __bf16* p, int ldb, int lane) {
  int n = lane & 15, kh = lane >> 4;
  const __bf16* r = p + (size_t)n * ldb + kh * 16;
  v16bf b;
#pragma unroll
  for (int i = 0; i < 16; ++i) b[i] = r[i];
  return b;
}

__device__ __forceinline__ v8f wmma_bf16(v16bf a, v16bf b, v8f c) {
  return __builtin_amdgcn_wmma_f32_16x16x32_bf16(false, a, false, b, (short)0, c,
                                                 false, false);
}

// --- fp32 -> bf16 conversion (weights) ---------------------------------------
__global__ void cvt_k(const float* __restrict__ s, __bf16* __restrict__ d, int n) {
  int i = blockIdx.x * blockDim.x + threadIdx.x;
  int stride = gridDim.x * blockDim.x;
  for (; i < n; i += stride) d[i] = (__bf16)s[i];
}

// --- residual-add + LayerNorm (also emits bf16 copy) -------------------------
__global__ __launch_bounds__(256) void add_ln_k(
    const float* __restrict__ x1, const float* __restrict__ x2, int bcast,
    const float* __restrict__ g, const float* __restrict__ bta,
    float* __restrict__ hout, __bf16* __restrict__ hb) {
  __shared__ float red[256];
  int row = blockIdx.x, tid = threadIdx.x;
  size_t o1 = (size_t)row * DD;
  size_t o2 = (size_t)(bcast ? (row % SS) : row) * DD;
  float val[3];
  float s = 0.f;
#pragma unroll
  for (int i = 0; i < 3; ++i) {
    int c = tid + i * 256;
    val[i] = x1[o1 + c] + x2[o2 + c];
    s += val[i];
  }
  red[tid] = s; __syncthreads();
  for (int off = 128; off > 0; off >>= 1) {
    if (tid < off) red[tid] += red[tid + off];
    __syncthreads();
  }
  float mean = red[0] * (1.0f / DD);
  __syncthreads();
  float s2 = 0.f;
#pragma unroll
  for (int i = 0; i < 3; ++i) { float dv = val[i] - mean; s2 += dv * dv; }
  red[tid] = s2; __syncthreads();
  for (int off = 128; off > 0; off >>= 1) {
    if (tid < off) red[tid] += red[tid + off];
    __syncthreads();
  }
  float rs = rsqrtf(red[0] * (1.0f / DD) + 1e-12f);
#pragma unroll
  for (int i = 0; i < 3; ++i) {
    int c = tid + i * 256;
    float o = (val[i] - mean) * rs * g[c] + bta[c];
    hout[o1 + c] = o;
    hb[o1 + c] = (__bf16)o;
  }
}

// --- WMMA GEMM: C[M,N] = (A_bf16[M,K] @ B_bf16[K,N] + bias) * scale ----------
// 128x128 tile per block; 8 waves of 32x64 each (2x4 wmma accumulators).
// A tile streamed with async global->LDS DMA (ASYNCcnt); B tile transposed
// through VGPRs into column-major LDS for contiguous B-fragment reads.
template <bool OUT_BF16, bool GELU>
__global__ __launch_bounds__(256) void gemm_k(
    const __bf16* __restrict__ A, const __bf16* __restrict__ Bw,
    const float* __restrict__ bias, float* __restrict__ Cf,
    __bf16* __restrict__ Cb, int M, int N, int K, float scale) {
  __shared__ __bf16 sA[128][40];  // 128 rows x 32 K  (pad 8)
  __shared__ __bf16 sB[128][40];  // 128 cols x 32 K  (transposed, pad 8)
  const int tid = threadIdx.x;
  const int lane = tid & 31, wave = tid >> 5;
  const int wr = wave >> 1, wc = wave & 1;
  const int row0 = blockIdx.y * 128, col0 = blockIdx.x * 128;
  const int kh = lane >> 4, cl = lane & 15;

  v8f zero = {0, 0, 0, 0, 0, 0, 0, 0};
  v8f acc[2][4];
#pragma unroll
  for (int i = 0; i < 2; ++i)
#pragma unroll
    for (int j = 0; j < 4; ++j) acc[i][j] = zero;

  for (int kb0 = 0; kb0 < K; kb0 += 32) {
    // A tile: async DMA straight into LDS (2 x 16B per thread)
#pragma unroll
    for (int it = 0; it < 2; ++it) {
      int e = (tid + it * 256) * 8;
      int r = e >> 5, c = e & 31;
      async_ld_lds_b128(lds_off(&sA[r][c]),
                        (unsigned long long)(uintptr_t)(A + (size_t)(row0 + r) * K +
                                                        kb0 + c));
    }
    // B tile: load rows, transpose into column-major LDS
#pragma unroll
    for (int it = 0; it < 2; ++it) {
      int k = (tid >> 4) + it * 16;
      int cb = (tid & 15) * 8;
      const __bf16* src = Bw + (size_t)(kb0 + k) * N + col0 + cb;
#pragma unroll
      for (int j = 0; j < 8; ++j) sB[cb + j][k] = src[j];
    }
    wait_async0();
    __syncthreads();

    v16bf af[2];
#pragma unroll
    for (int mt = 0; mt < 2; ++mt)
      af[mt] = load_a_frag(&sA[wr * 32 + mt * 16][0], 40, lane);
#pragma unroll
    for (int nt = 0; nt < 4; ++nt) {
      v16bf bf = load_b_frag(&sB[wc * 64 + nt * 16][0], 40, lane);
#pragma unroll
      for (int mt = 0; mt < 2; ++mt) acc[mt][nt] = wmma_bf16(af[mt], bf, acc[mt][nt]);
    }
    __syncthreads();
  }

  // epilogue: bias, scale, optional exact GELU, fp32 or bf16 store
#pragma unroll
  for (int mt = 0; mt < 2; ++mt)
#pragma unroll
    for (int nt = 0; nt < 4; ++nt) {
      int cN = col0 + wc * 64 + nt * 16 + cl;
      float bs = bias[cN];
#pragma unroll
      for (int v = 0; v < 8; ++v) {
        int r = row0 + wr * 32 + mt * 16 + kh * 8 + v;
        float val = (acc[mt][nt][v] + bs) * scale;
        if (GELU) val = 0.5f * val * (1.0f + erff(val * 0.70710678118f));
        if (OUT_BF16) Cb[(size_t)r * N + cN] = (__bf16)val;
        else          Cf[(size_t)r * N + cN] = val;
      }
    }
}

// --- sliding-window attention, flash-style online softmax --------------------
// grid = B*H*NC blocks; each wave owns 16 query rows in registers. The V chunk
// (shared by all 8 waves) is staged block-wide into LDS with async DMA,
// overlapped with the Q@K^T WMMAs.
__global__ __launch_bounds__(256) void attn_k(
    const __bf16* __restrict__ q, const __bf16* __restrict__ kM,
    const __bf16* __restrict__ vM, __bf16* __restrict__ o) {
  __shared__ __bf16 sP[8][16 * 136];  // per-wave P staging (C->A layout swap)
  __shared__ __bf16 sV[128][72];      // shared V chunk (128 keys x 64 dh, pad 8)
  int bid = blockIdx.x;
  int c = bid % NCH;
  int h = (bid / NCH) % HH;
  int b = bid / (NCH * HH);
  int tid = threadIdx.x, lane = tid & 31, wave = tid >> 5;
  int kh = lane >> 4, ln = lane & 15;
  int qr0 = wave * 16;
  size_t bbase = (size_t)b * SS;
  int hoff = h * DHD;

  // Q fragments (q already scaled by DH^-0.5 in projection)
  v16bf qf[2];
  {
    const __bf16* qb = q + (bbase + (size_t)c * WW + qr0) * DD + hoff;
    qf[0] = load_a_frag(qb, DD, lane);
    qf[1] = load_a_frag(qb + 32, DD, lane);
  }
  v8f zero = {0, 0, 0, 0, 0, 0, 0, 0};
  float mrow[8], lrow[8];
#pragma unroll
  for (int v = 0; v < 8; ++v) { mrow[v] = -3.0e38f; lrow[v] = 0.f; }
  v8f ot[4];
#pragma unroll
  for (int t = 0; t < 4; ++t) ot[t] = zero;

  for (int kc = 0; kc < 3; ++kc) {
    int kt0 = c * WW - WW + kc * WW;

    // kick off async V-chunk staging (4 x 16B per thread), overlaps with Q@K^T
#pragma unroll
    for (int it = 0; it < 4; ++it) {
      int e = (tid + it * 256) * 8;
      int r = e >> 6, col = e & 63;
      int vt = kt0 + r;
      int vtc = imin(imax(vt, 0), SS - 1);
      async_ld_lds_b128(
          lds_off(&sV[r][col]),
          (unsigned long long)(uintptr_t)(vM + (bbase + vtc) * DD + hoff + col));
    }

    // S = Q @ K^T for this 128-key chunk (8 n-tiles), K frags from global
    v8f sc[8];
#pragma unroll
    for (int knt = 0; knt < 8; ++knt) {
      int kt = kt0 + knt * 16 + ln;
      int ktc = imin(imax(kt, 0), SS - 1);
      const __bf16* kp = kM + (bbase + ktc) * DD + hoff + kh * 16;
      v16bf b0, b1;
#pragma unroll
      for (int i = 0; i < 16; ++i) { b0[i] = kp[i]; b1[i] = kp[32 + i]; }
      v8f s = wmma_bf16(qf[0], b0, zero);
      s = wmma_bf16(qf[1], b1, s);
      sc[knt] = s;
    }
    // band + bounds mask (-1e9 like reference), then online softmax
    float mloc[8];
#pragma unroll
    for (int v = 0; v < 8; ++v) mloc[v] = -3.0e38f;
#pragma unroll
    for (int knt = 0; knt < 8; ++knt) {
      int kpos = kt0 + knt * 16 + ln;
#pragma unroll
      for (int v = 0; v < 8; ++v) {
        int qpos = c * WW + qr0 + kh * 8 + v;
        int d = kpos - qpos;
        bool valid = (kpos >= 0) & (kpos < SS) & (d <= WW) & (d >= -WW);
        float s = valid ? sc[knt][v] : -1.0e9f;
        sc[knt][v] = s;
        mloc[v] = fmaxf(mloc[v], s);
      }
    }
#pragma unroll
    for (int v = 0; v < 8; ++v) {
#pragma unroll
      for (int off = 1; off < 16; off <<= 1)
        mloc[v] = fmaxf(mloc[v], __shfl_xor(mloc[v], off, 32));
    }
    float corr[8];
#pragma unroll
    for (int v = 0; v < 8; ++v) {
      float mn = fmaxf(mrow[v], mloc[v]);
      corr[v] = expf(mrow[v] - mn);
      mrow[v] = mn;
    }
    float rsum[8];
#pragma unroll
    for (int v = 0; v < 8; ++v) rsum[v] = 0.f;
#pragma unroll
    for (int knt = 0; knt < 8; ++knt)
#pragma unroll
      for (int v = 0; v < 8; ++v) {
        float p = expf(sc[knt][v] - mrow[v]);
        sc[knt][v] = p;
        rsum[v] += p;
      }
#pragma unroll
    for (int v = 0; v < 8; ++v) {
#pragma unroll
      for (int off = 1; off < 16; off <<= 1)
        rsum[v] += __shfl_xor(rsum[v], off, 32);
      lrow[v] = lrow[v] * corr[v] + rsum[v];
    }
#pragma unroll
    for (int t = 0; t < 4; ++t)
#pragma unroll
      for (int v = 0; v < 8; ++v) ot[t][v] *= corr[v];

    // stage P (C-layout) to LDS, reload as A-fragments
    __bf16* sw = &sP[wave][0];
#pragma unroll
    for (int knt = 0; knt < 8; ++knt)
#pragma unroll
      for (int v = 0; v < 8; ++v)
        sw[(kh * 8 + v) * 136 + knt * 16 + ln] = (__bf16)sc[knt][v];

    wait_async0();   // own V DMAs done
    __syncthreads(); // everyone's V DMAs + P stores visible

    v16bf pf[4];
#pragma unroll
    for (int f = 0; f < 4; ++f) pf[f] = load_a_frag(sw + f * 32, 136, lane);
    // O += P @ V   (V fragments from the shared LDS chunk)
#pragma unroll
    for (int t = 0; t < 4; ++t)
#pragma unroll
      for (int f = 0; f < 4; ++f) {
        v16bf vf;
#pragma unroll
        for (int i = 0; i < 16; ++i)
          vf[i] = sV[f * 32 + kh * 16 + i][t * 16 + ln];
        ot[t] = wmma_bf16(pf[f], vf, ot[t]);
      }
    __syncthreads(); // protect sV/sP overwrite next iteration
  }
  // normalize and write bf16 output
#pragma unroll
  for (int t = 0; t < 4; ++t)
#pragma unroll
    for (int v = 0; v < 8; ++v) {
      size_t tok = bbase + (size_t)c * WW + qr0 + kh * 8 + v;
      o[tok * DD + hoff + t * 16 + ln] = (__bf16)(ot[t][v] / lrow[v]);
    }
}

// -----------------------------------------------------------------------------
extern "C" void kernel_launch(void* const* d_in, const int* in_sizes, int n_in,
                              void* d_out, int out_size, void* d_ws,
                              size_t ws_size, hipStream_t stream) {
  (void)in_sizes; (void)n_in; (void)out_size; (void)ws_size;
  const float* x    = (const float*)d_in[0];
  const float* pos  = (const float*)d_in[1];
  const float* eg   = (const float*)d_in[2];
  const float* ebi  = (const float*)d_in[3];
  const float* Wq   = (const float*)d_in[4];
  const float* bq   = (const float*)d_in[5];
  const float* Wk   = (const float*)d_in[6];
  const float* bk   = (const float*)d_in[7];
  const float* Wv   = (const float*)d_in[8];
  const float* bv   = (const float*)d_in[9];
  const float* Wo   = (const float*)d_in[10];
  const float* bo   = (const float*)d_in[11];
  const float* ln1g = (const float*)d_in[12];
  const float* ln1b = (const float*)d_in[13];
  const float* W1   = (const float*)d_in[14];
  const float* b1   = (const float*)d_in[15];
  const float* W2   = (const float*)d_in[16];
  const float* b2   = (const float*)d_in[17];
  const float* ln2g = (const float*)d_in[18];
  const float* ln2b = (const float*)d_in[19];

  char* base = (char*)d_ws;
  size_t off = 0;
  auto alloc = [&](size_t bytes) -> char* {
    char* p = base + off;
    off += (bytes + 255) & ~(size_t)255;
    return p;
  };
  float*  h  = (float*)alloc((size_t)TT * DD * 4);
  float*  y  = (float*)alloc((size_t)TT * DD * 4);
  __bf16* hb = (__bf16*)alloc((size_t)TT * DD * 2);
  char* U = alloc((size_t)TT * FF * 2);  // aliases qb/kb/vb/ab vs fb
  __bf16* qb = (__bf16*)U;
  __bf16* kb = (__bf16*)(U + (size_t)TT * DD * 2);
  __bf16* vb = (__bf16*)(U + (size_t)TT * DD * 2 * 2);
  __bf16* ab = (__bf16*)(U + (size_t)TT * DD * 2 * 3);
  __bf16* fb = (__bf16*)U;
  __bf16* wqb = (__bf16*)alloc((size_t)DD * DD * 2);
  __bf16* wkb = (__bf16*)alloc((size_t)DD * DD * 2);
  __bf16* wvb = (__bf16*)alloc((size_t)DD * DD * 2);
  __bf16* wob = (__bf16*)alloc((size_t)DD * DD * 2);
  __bf16* w1b = (__bf16*)alloc((size_t)DD * FF * 2);
  __bf16* w2b = (__bf16*)alloc((size_t)FF * DD * 2);

  const float SCALE_Q = 0.125f;  // DH^-0.5

  // embeddings: h = LN(x + pos)
  add_ln_k<<<TT, 256, 0, stream>>>(x, pos, 1, eg, ebi, h, hb);

  dim3 gD(DD / 128, TT / 128);
  dim3 gF(FF / 128, TT / 128);

  for (int l = 0; l < LL; ++l) {
    cvt_k<<<2304, 256, 0, stream>>>(Wq + (size_t)l * DD * DD, wqb, DD * DD);
    cvt_k<<<2304, 256, 0, stream>>>(Wk + (size_t)l * DD * DD, wkb, DD * DD);
    cvt_k<<<2304, 256, 0, stream>>>(Wv + (size_t)l * DD * DD, wvb, DD * DD);
    cvt_k<<<2304, 256, 0, stream>>>(Wo + (size_t)l * DD * DD, wob, DD * DD);
    cvt_k<<<2304, 256, 0, stream>>>(W1 + (size_t)l * DD * FF, w1b, DD * FF);
    cvt_k<<<2304, 256, 0, stream>>>(W2 + (size_t)l * FF * DD, w2b, FF * DD);

    gemm_k<true, false><<<gD, 256, 0, stream>>>(hb, wqb, bq + l * DD, nullptr, qb,
                                                TT, DD, DD, SCALE_Q);
    gemm_k<true, false><<<gD, 256, 0, stream>>>(hb, wkb, bk + l * DD, nullptr, kb,
                                                TT, DD, DD, 1.0f);
    gemm_k<true, false><<<gD, 256, 0, stream>>>(hb, wvb, bv + l * DD, nullptr, vb,
                                                TT, DD, DD, 1.0f);

    attn_k<<<BB * HH * NCH, 256, 0, stream>>>(qb, kb, vb, ab);

    gemm_k<false, false><<<gD, 256, 0, stream>>>(ab, wob, bo + l * DD, y, nullptr,
                                                 TT, DD, DD, 1.0f);
    add_ln_k<<<TT, 256, 0, stream>>>(h, y, 0, ln1g + l * DD, ln1b + l * DD, h, hb);

    gemm_k<true, true><<<gF, 256, 0, stream>>>(hb, w1b, b1 + l * FF, nullptr, fb,
                                               TT, FF, DD, 1.0f);
    gemm_k<false, false><<<gD, 256, 0, stream>>>(fb, w2b, b2 + l * DD, y, nullptr,
                                                 TT, DD, FF, 1.0f);
    float* hdst = (l == LL - 1) ? (float*)d_out : h;
    add_ln_k<<<TT, 256, 0, stream>>>(h, y, 0, ln2g + l * DD, ln2b + l * DD, hdst, hb);
  }
}
